// MultiHeadAttention_14516989461009
// MI455X (gfx1250) — compile-verified, matrix operand order checked by reading, not run on hardware
//
#include <hip/hip_runtime.h>
#include <hip/hip_bf16.h>

// ---------------------------------------------------------------------------
// MHA forward for MI455X (gfx1250): bf16 WMMA (v_wmma_f32_16x16x32_bf16)
// pipeline: cvt->bf16, Q/K/V projections, flash attention, output projection.
// GEMMs use async global->LDS copies (ASYNCcnt) for the shared weight tile,
// double-buffered; wave tile 32Mx64N with 64-K stages = 16 WMMAs per barrier.
// ---------------------------------------------------------------------------

typedef __attribute__((ext_vector_type(16))) __bf16 bf16x16;
typedef __attribute__((ext_vector_type(8)))  __bf16 bf16x8;
typedef __attribute__((ext_vector_type(8)))  float  f32x8;

static constexpr int Bsz   = 2;
static constexpr int Ssz   = 2048;
static constexpr int Dm    = 1024;   // d_model
static constexpr int Hh    = 16;     // heads
static constexpr int Dh    = 64;     // head dim
static constexpr int Mrows = Bsz * Ssz;       // 4096

union BF16x16U { bf16x16 v; bf16x8 h[2]; };

// Load one WMMA A/B fragment (16 bf16 per lane) from a row-major matrix.
// Lane holds row/col (lane&15); K packing: low half -> K {k..k+7, k+16..k+23},
// high half -> K {k+8..k+15, k+24..k+31}.
template <typename PtrT>
__device__ __forceinline__ bf16x16 load_frag(PtrT row, int k, int half) {
    BF16x16U u;
    const int k0 = k + half * 8;
    u.h[0] = *reinterpret_cast<const bf16x8*>(row + k0);
    u.h[1] = *reinterpret_cast<const bf16x8*>(row + k0 + 16);
    return u.v;
}

__device__ __forceinline__ f32x8 wmma_bf16(bf16x16 a, bf16x16 b, f32x8 c) {
    return __builtin_amdgcn_wmma_f32_16x16x32_bf16(
        /*neg_a=*/false, a, /*neg_b=*/false, b,
        /*c_mod=*/(short)0, c, /*reuse_a=*/false, /*reuse_b=*/false);
}

// Async copy of 16 bytes global -> LDS, tracked by ASYNCcnt.
__device__ __forceinline__ void async_copy_b128(uint32_t lds_off, const void* gsrc) {
    asm volatile("global_load_async_to_lds_b128 %0, %1, off"
                 :: "v"(lds_off), "v"(gsrc) : "memory");
}
__device__ __forceinline__ void wait_async0() {
    asm volatile("s_wait_asynccnt 0x0" ::: "memory");
}

// ------------------------- fp32 -> bf16 conversion -------------------------
struct BF4 { __bf16 a, b, c, d; };

__global__ void cvt_bf16_kernel(const float* __restrict__ in,
                                __bf16* __restrict__ out, int n4) {
    int i = blockIdx.x * blockDim.x + threadIdx.x;
    if (i < n4) {
        float4 f = reinterpret_cast<const float4*>(in)[i];
        BF4 o;
        o.a = (__bf16)f.x; o.b = (__bf16)f.y;
        o.c = (__bf16)f.z; o.d = (__bf16)f.w;
        reinterpret_cast<BF4*>(out)[i] = o;
    }
}

// ------------------------- generic WMMA GEMM -------------------------------
// C[m,n] = sum_k A[m,k] * W[n,k] + bias[n]
//   mode 0: f32 out, plain [M, Dm]               (final projection)
//   mode 1: bf16 out, split heads [B,H,S,Dh]     (Q, K)
//   mode 2: bf16 out, split heads T [B,H,Dh,S]   (V transposed)
// Block = 256 threads (8 waves); block tile 256M x 64N; wave tile 32M x 64N.
// Weight tile 64N x 64K staged in LDS via async copies, double buffered.
__global__ __launch_bounds__(256)
void gemm_wmma_kernel(const __bf16* __restrict__ A,
                      const __bf16* __restrict__ W,
                      const float* __restrict__ bias,
                      float* __restrict__ outF,
                      __bf16* __restrict__ outB,
                      int mode) {
    constexpr int LDW = 72;  // padded LDS row (elems): 144B stride -> conflict-free b128
    __shared__ __align__(16) __bf16 ldsW[2][64 * LDW];   // 18 KB

    const int tid  = threadIdx.x;
    const int lane = tid & 31;
    const int wave = tid >> 5;
    const int l16  = lane & 15;
    const int half = lane >> 4;

    const int mBase = blockIdx.x * 256 + wave * 32;   // grid.x = 4096/256
    const int nBase = blockIdx.y * 64;                // grid.y = 1024/64

    // Async-copy assignment: thread copies 2x16B of W row (tid>>2),
    // K-chunks (tid&3)*8 and (tid&3)*8 + 32 within the 64-K stage.
    const int cRow = tid >> 2;                        // 0..63
    const int cK   = (tid & 3) * 8;                   // 0,8,16,24
    const __bf16* wsrc = W + (size_t)(nBase + cRow) * Dm + cK;
    uint32_t ldstA[2], ldstB[2];
    ldstA[0] = (uint32_t)(uintptr_t)&ldsW[0][cRow * LDW + cK];
    ldstB[0] = (uint32_t)(uintptr_t)&ldsW[0][cRow * LDW + cK + 32];
    ldstA[1] = (uint32_t)(uintptr_t)&ldsW[1][cRow * LDW + cK];
    ldstB[1] = (uint32_t)(uintptr_t)&ldsW[1][cRow * LDW + cK + 32];

    // prime buffer 0 (k = 0)
    async_copy_b128(ldstA[0], wsrc);
    async_copy_b128(ldstB[0], wsrc + 32);

    const __bf16* Arow0 = A + (size_t)(mBase +  0 + l16) * Dm;
    const __bf16* Arow1 = A + (size_t)(mBase + 16 + l16) * Dm;

    bf16x16 a_cur[4];   // [Msub0 k0, Msub0 k32, Msub1 k0, Msub1 k32]
    a_cur[0] = load_frag(Arow0, 0, half);
    a_cur[1] = load_frag(Arow0, 32, half);
    a_cur[2] = load_frag(Arow1, 0, half);
    a_cur[3] = load_frag(Arow1, 32, half);

    f32x8 accA[4], accB[4];
    #pragma unroll
    for (int t = 0; t < 4; ++t)
        #pragma unroll
        for (int i = 0; i < 8; ++i) { accA[t][i] = 0.f; accB[t][i] = 0.f; }

    wait_async0();
    __syncthreads();

    for (int k = 0; k < Dm; k += 64) {
        const int buf = (k >> 6) & 1;
        bf16x16 a_nxt[4];
        #pragma unroll
        for (int i = 0; i < 4; ++i) a_nxt[i] = a_cur[i];
        if (k + 64 < Dm) {
            a_nxt[0] = load_frag(Arow0, k + 64, half);    // prefetch A
            a_nxt[1] = load_frag(Arow0, k + 96, half);
            a_nxt[2] = load_frag(Arow1, k + 64, half);
            a_nxt[3] = load_frag(Arow1, k + 96, half);
            async_copy_b128(ldstA[buf ^ 1], wsrc + k + 64);   // prefetch W
            async_copy_b128(ldstB[buf ^ 1], wsrc + k + 96);
        }
        const __bf16* wt = &ldsW[buf][0];
        #pragma unroll
        for (int t = 0; t < 4; ++t) {
            const __bf16* wrow = wt + (size_t)(t * 16 + l16) * LDW;
            const bf16x16 b0 = load_frag(wrow, 0, half);
            const bf16x16 b1 = load_frag(wrow, 32, half);
            accA[t] = wmma_bf16(a_cur[0], b0, accA[t]);
            accA[t] = wmma_bf16(a_cur[1], b1, accA[t]);
            accB[t] = wmma_bf16(a_cur[2], b0, accB[t]);
            accB[t] = wmma_bf16(a_cur[3], b1, accB[t]);
        }
        #pragma unroll
        for (int i = 0; i < 4; ++i) a_cur[i] = a_nxt[i];
        wait_async0();          // my copies into next buffer done
        __syncthreads();        // everyone's copies done; tile reads retired
    }

    #pragma unroll
    for (int ms = 0; ms < 2; ++ms) {
        #pragma unroll
        for (int t = 0; t < 4; ++t) {
            const int n  = nBase + t * 16 + l16;
            const float bv = bias[n];
            #pragma unroll
            for (int i = 0; i < 8; ++i) {
                const int m = mBase + ms * 16 + i + 8 * half;
                const float v = (ms ? accB[t][i] : accA[t][i]) + bv;
                if (mode == 0) {
                    outF[(size_t)m * Dm + n] = v;
                } else {
                    const int b = m >> 11;        // m / 2048
                    const int s = m & 2047;
                    const int h = n >> 6;         // n / 64
                    const int d = n & 63;
                    if (mode == 1)
                        outB[(((size_t)b * Hh + h) * Ssz + s) * Dh + d] = (__bf16)v;
                    else
                        outB[(((size_t)b * Hh + h) * Dh + d) * Ssz + s] = (__bf16)v;
                }
            }
        }
    }
}

// ------------------------- flash attention ---------------------------------
// Qh, Kh: [B,H,S,Dh] bf16; Vt: [B,H,Dh,S] bf16; ctx out: [B,S,Dm] bf16.
// One wave handles a 16-query tile over all S keys (blocks of 32 keys).
__global__ __launch_bounds__(128)
void attn_wmma_kernel(const __bf16* __restrict__ Qh,
                      const __bf16* __restrict__ Kh,
                      const __bf16* __restrict__ Vt,
                      __bf16* __restrict__ ctx) {
    __shared__ __align__(16) __bf16 pT[4][16 * 32];   // wave-private 16x32 P tiles

    const int lane = threadIdx.x & 31;
    const int wave = threadIdx.x >> 5;
    const int l16  = lane & 15;
    const int half = lane >> 4;

    const int tile  = blockIdx.x * 4 + wave;     // 4096 tiles total
    const int bh    = tile >> 7;                 // 128 q-tiles per (b,h)
    const int qBase = (tile & 127) << 4;

    const __bf16* Qb = Qh + (size_t)bh * Ssz * Dh;
    const __bf16* Kb = Kh + (size_t)bh * Ssz * Dh;
    const __bf16* Vb = Vt + (size_t)bh * Dh * Ssz;
    __bf16* pt = &pT[wave][0];

    const __bf16* qrow = Qb + (size_t)(qBase + l16) * Dh;
    const bf16x16 qa0 = load_frag(qrow, 0, half);
    const bf16x16 qa1 = load_frag(qrow, 32, half);

    f32x8 cacc[4];
    float mrun[8], lrun[8];
    #pragma unroll
    for (int i = 0; i < 8; ++i) {
        mrun[i] = -1e30f; lrun[i] = 0.f;
        cacc[0][i]=0.f; cacc[1][i]=0.f; cacc[2][i]=0.f; cacc[3][i]=0.f;
    }

    for (int j = 0; j < Ssz; j += 32) {
        // scores: 16 queries x 32 keys, K-dim = Dh = 64 (two WMMA steps)
        f32x8 s0, s1;
        #pragma unroll
        for (int i = 0; i < 8; ++i) { s0[i]=0.f; s1[i]=0.f; }
        const __bf16* kr0 = Kb + (size_t)(j + l16) * Dh;
        const __bf16* kr1 = Kb + (size_t)(j + 16 + l16) * Dh;
        s0 = wmma_bf16(qa0, load_frag(kr0, 0,  half), s0);
        s0 = wmma_bf16(qa1, load_frag(kr0, 32, half), s0);
        s1 = wmma_bf16(qa0, load_frag(kr1, 0,  half), s1);
        s1 = wmma_bf16(qa1, load_frag(kr1, 32, half), s1);

        // online softmax per row (rows of VGPR i live in one 16-lane half)
        #pragma unroll
        for (int i = 0; i < 8; ++i) {
            const float sa = s0[i] * 0.125f;   // 1/sqrt(64)
            const float sb = s1[i] * 0.125f;
            float mx = fmaxf(sa, sb);
            #pragma unroll
            for (int off = 1; off < 16; off <<= 1)
                mx = fmaxf(mx, __shfl_xor(mx, off, 32));
            const float mnew = fmaxf(mrun[i], mx);
            const float pa = __expf(sa - mnew);
            const float pb = __expf(sb - mnew);
            float rs = pa + pb;
            #pragma unroll
            for (int off = 1; off < 16; off <<= 1)
                rs += __shfl_xor(rs, off, 32);
            const float corr = __expf(mrun[i] - mnew);
            lrun[i] = lrun[i] * corr + rs;
            mrun[i] = mnew;
            cacc[0][i] *= corr; cacc[1][i] *= corr;
            cacc[2][i] *= corr; cacc[3][i] *= corr;
            // scatter P (C layout) into LDS row-major [16][32]
            const int m = i + 8 * half;
            pt[m * 32 + l16]      = (__bf16)pa;
            pt[m * 32 + 16 + l16] = (__bf16)pb;
        }
        asm volatile("s_wait_dscnt 0x0" ::: "memory");  // wave-local LDS RAW

        // reload P in A-fragment layout, then ctx += P @ V (K-dim = 32 keys)
        const bf16x16 pfrag = load_frag(pt + l16 * 32, 0, half);
        #pragma unroll
        for (int t = 0; t < 4; ++t) {
            const __bf16* vrow = Vb + (size_t)(t * 16 + l16) * Ssz;
            cacc[t] = wmma_bf16(pfrag, load_frag(vrow, j, half), cacc[t]);
        }
    }

    // normalize and store merged-head ctx [B,S,Dm]
    const int b = bh >> 4;
    const int h = bh & 15;
    #pragma unroll
    for (int t = 0; t < 4; ++t) {
        #pragma unroll
        for (int i = 0; i < 8; ++i) {
            const int q = qBase + i + 8 * half;
            const int d = h * Dh + t * 16 + l16;
            const float v = cacc[t][i] / lrun[i];
            ctx[((size_t)b * Ssz + q) * Dm + d] = (__bf16)v;
        }
    }
}

// ------------------------- host-side orchestration -------------------------
extern "C" void kernel_launch(void* const* d_in, const int* in_sizes, int n_in,
                              void* d_out, int out_size, void* d_ws, size_t ws_size,
                              hipStream_t stream) {
    const float* q_in = (const float*)d_in[0];
    const float* k_in = (const float*)d_in[1];
    const float* v_in = (const float*)d_in[2];
    const float* Wq   = (const float*)d_in[3];
    const float* bq   = (const float*)d_in[4];
    const float* Wk   = (const float*)d_in[5];
    const float* bk   = (const float*)d_in[6];
    const float* Wv   = (const float*)d_in[7];
    const float* bv   = (const float*)d_in[8];
    const float* Wo   = (const float*)d_in[9];
    const float* bo   = (const float*)d_in[10];

    char* ws = (char*)d_ws;
    const size_t MB = 1ull << 20;
    __bf16* xq  = (__bf16*)(ws +  0 * MB);   // [4096,1024]  8MB
    __bf16* xk  = (__bf16*)(ws +  8 * MB);
    __bf16* xv  = (__bf16*)(ws + 16 * MB);
    __bf16* wqb = (__bf16*)(ws + 24 * MB);   // [1024,1024]  2MB each
    __bf16* wkb = (__bf16*)(ws + 26 * MB);
    __bf16* wvb = (__bf16*)(ws + 28 * MB);
    __bf16* wob = (__bf16*)(ws + 30 * MB);
    __bf16* Qh  = (__bf16*)(ws + 32 * MB);   // [B,H,S,Dh]   8MB
    __bf16* Kh  = (__bf16*)(ws + 40 * MB);
    __bf16* Vt  = (__bf16*)(ws + 48 * MB);   // [B,H,Dh,S]   8MB
    __bf16* ctx = (__bf16*)(ws + 56 * MB);   // [B,S,Dm]     8MB

    const int nInp4 = (Mrows * Dm) / 4;      // 1,048,576
    const int nW4   = (Dm * Dm) / 4;         //   262,144
    const int cvtB  = 256;
    cvt_bf16_kernel<<<(nInp4 + cvtB - 1) / cvtB, cvtB, 0, stream>>>(q_in, xq, nInp4);
    cvt_bf16_kernel<<<(nInp4 + cvtB - 1) / cvtB, cvtB, 0, stream>>>(k_in, xk, nInp4);
    cvt_bf16_kernel<<<(nInp4 + cvtB - 1) / cvtB, cvtB, 0, stream>>>(v_in, xv, nInp4);
    cvt_bf16_kernel<<<(nW4   + cvtB - 1) / cvtB, cvtB, 0, stream>>>(Wq, wqb, nW4);
    cvt_bf16_kernel<<<(nW4   + cvtB - 1) / cvtB, cvtB, 0, stream>>>(Wk, wkb, nW4);
    cvt_bf16_kernel<<<(nW4   + cvtB - 1) / cvtB, cvtB, 0, stream>>>(Wv, wvb, nW4);
    cvt_bf16_kernel<<<(nW4   + cvtB - 1) / cvtB, cvtB, 0, stream>>>(Wo, wob, nW4);

    const dim3 gGrid(Mrows / 256, Dm / 64);   // (16, 16)
    gemm_wmma_kernel<<<gGrid, 256, 0, stream>>>(xq, wqb, bq, nullptr, Qh, 1);
    gemm_wmma_kernel<<<gGrid, 256, 0, stream>>>(xk, wkb, bk, nullptr, Kh, 1);
    gemm_wmma_kernel<<<gGrid, 256, 0, stream>>>(xv, wvb, bv, nullptr, Vt, 2);

    attn_wmma_kernel<<<(Bsz * Hh * (Ssz / 16)) / 4, 128, 0, stream>>>(Qh, Kh, Vt, ctx);

    gemm_wmma_kernel<<<gGrid, 256, 0, stream>>>(ctx, wob, bo, (float*)d_out, nullptr, 0);
}
